// V4Attention_13838384627854
// MI455X (gfx1250) — matrix operand order repaired and checked by reading, not compiled, not run on hardware
//
#include <hip/hip_runtime.h>

#define HEADD 128
#define HKV 8
#define GQ 4
#define HQn 32
#define WIN 512
#define TILE_K 32
#define NWAVES 4
#define VROW 129            // LDS row stride in floats (128 + 1 pad from TDM pad feature)
#define VSTAGE_FLOATS (TILE_K * VROW)

typedef __attribute__((ext_vector_type(16))) __bf16 bfrag_t; // WMMA A/B operand
typedef __attribute__((ext_vector_type(8)))  float  cfrag_t; // WMMA C/D operand
typedef __attribute__((ext_vector_type(4)))  float  f4_t;
typedef unsigned int u32x4 __attribute__((ext_vector_type(4)));
typedef int          i32x4 __attribute__((ext_vector_type(4)));
typedef int          i32x8 __attribute__((ext_vector_type(8)));

__device__ __forceinline__ float wave_bcast0(float v) { return __shfl(v, 0, 32); }

// Issue a TDM 2D tile load: 32 rows x 128 f32, row stride strideElems, into LDS
// at ldsAddr with a 1-dword pad after every 128 dwords (row stride 129 floats).
__device__ __forceinline__ void tdm_load_v_tile(unsigned long long gaddr,
                                                unsigned ldsAddr,
                                                unsigned strideElems) {
  // D# group 0: count=1 | lds_addr | global_addr(57b) | type=2
  u32x4 g0;
  g0[0] = 1u;                                        // count=1, user mode, no gather
  g0[1] = ldsAddr;                                   // LDS byte address
  g0[2] = (unsigned)(gaddr & 0xffffffffu);           // global_addr[31:0]
  g0[3] = (unsigned)(gaddr >> 32) | 0x80000000u;     // global_addr[56:32] | type=2<<30
  // D# group 1
  i32x8 g1;
  g1[0] = (int)((2u << 16)        // data_size = 2 (4 bytes)
              | (1u << 20)        // pad_enable
              | (6u << 22));      // pad_interval code 6 = every 128 dwords (+pad_amount code 0 = 1 dword)
  g1[1] = (int)((unsigned)(HEADD & 0xffff) << 16);            // tensor_dim0[15:0] (=128)
  g1[2] = (int)((HEADD >> 16) | ((TILE_K & 0xffff) << 16));   // tensor_dim0[31:16] | tensor_dim1[15:0]
  g1[3] = (int)((TILE_K >> 16) | ((unsigned)HEADD << 16));    // tensor_dim1[31:16] | tile_dim0 (=128)
  g1[4] = TILE_K;                                             // tile_dim1 (=32), tile_dim2 = 0 -> 2D
  g1[5] = (int)strideElems;                                   // tensor_dim0_stride[31:0]
  g1[6] = 0;                                                  // stride0[47:32] | stride1[15:0]
  g1[7] = 0;
  i32x4 gz = {0, 0, 0, 0};
#if __clang_major__ >= 23
  i32x8 gz8 = {0, 0, 0, 0, 0, 0, 0, 0};
  __builtin_amdgcn_tensor_load_to_lds(g0, g1, gz, gz, gz8, 0);
#else
  __builtin_amdgcn_tensor_load_to_lds(g0, g1, gz, gz, 0);
#endif
}

__global__ __launch_bounds__(128, 1)
void nsa_decode(const float* __restrict__ q,
                const float* __restrict__ raw,
                const float* __restrict__ sink,
                const float* __restrict__ comp,
                const int*   __restrict__ btab,
                float* __restrict__ out,
                int nc, int nbps, int winTok)
{
  const int bh   = blockIdx.x;
  const int b    = bh / HKV;
  const int kh   = bh % HKV;
  const int tid  = threadIdx.x;
  const int lane = tid & 31;
  const int wave = tid >> 5;
  const int mrow = lane & 15;
  const bool hi  = lane >= 16;

  __shared__ float lds_q[GQ][HEADD];
  __shared__ float lds_p[NWAVES][GQ][TILE_K];
  __shared__ float lds_o[NWAVES][GQ][HEADD];
  __shared__ float lds_m[NWAVES][GQ];
  __shared__ float lds_l[NWAVES][GQ];
  __shared__ __align__(16) float vstage[NWAVES][VSTAGE_FLOATS];  // TDM V staging

  // Load Q for this kv-head group, pre-multiplied by softmax scale.
  for (int i = tid; i < GQ * HEADD; i += blockDim.x) {
    int g = i >> 7, d = i & 127;
    lds_q[g][d] = q[((b * HQn) + kh * GQ + g) * HEADD + d] * 0.08838834764831845f;
  }
  __syncthreads();

  // Build Q A-fragments (16x32 bf16 per 32-dim chunk); rows >= GQ zero-filled.
  bfrag_t qa[4];
  #pragma unroll
  for (int c = 0; c < 4; ++c) {
    #pragma unroll
    for (int e = 0; e < 16; ++e) {
      int v = e >> 1, h = e & 1;
      int base = hi ? 8 : 0;
      int k = 2 * base + e;
      float val = (mrow < GQ) ? lds_q[mrow][c * 32 + k] : 0.0f;
      qa[c][e] = (__bf16)val;
    }
  }

  cfrag_t oacc[8];
  #pragma unroll
  for (int i = 0; i < 8; ++i)
    #pragma unroll
    for (int j = 0; j < 8; ++j) oacc[i][j] = 0.0f;

  float Mrun[GQ], Lrun[GQ];
  #pragma unroll
  for (int g = 0; g < GQ; ++g) { Mrun[g] = -3.0e38f; Lrun[g] = 0.0f; }

  const int ncTiles = nc / TILE_K;
  const int nTiles  = ncTiles + WIN / TILE_K;

  const float* vsp = &vstage[wave][0];
  const unsigned vstageAddr = (unsigned)(unsigned long long)vsp;  // LDS byte offset
  const int vbase = (hi ? 16 : 0) * VROW + mrow;                  // per-lane V gather base

  for (int t = wave; t < nTiles; t += NWAVES) {
    // A 32-key tile never crosses a paged block (tiles are 32-aligned in 64-token
    // blocks), so the block-table lookup and row stride are wave-uniform.
    const float* bp;
    long key0;           // element offset of key 0 of this tile (c=0 plane)
    int strideElems, voffElems;
    if (t < ncTiles) {
      key0 = ((long)(b * nc + t * TILE_K) * 2 * HKV + kh) * HEADD;   // (B, nc, 2, Hk, D)
      bp = comp; strideElems = 2 * HKV * HEADD; voffElems = HKV * HEADD;
    } else {
      int tok0 = winTok + (t - ncTiles) * TILE_K;
      int phys = btab[b * nbps + (tok0 >> 6)];
      key0 = (((long)phys * 2 * 64 + (tok0 & 63)) * HKV + kh) * HEADD; // (nb, 2, 64, Hk, D)
      bp = raw; strideElems = HKV * HEADD; voffElems = 64 * HKV * HEADD;
    }

    // Kick off the V tile DMA (TDM, async, tracked by TENSORcnt) — overlaps
    // with the K loads, QK WMMAs and softmax below.
    tdm_load_v_tile((unsigned long long)(const void*)bp + 4ull * (key0 + voffElems),
                    vstageAddr, (unsigned)strideElems);

    // K fragment pointers: lane (l%16) covers key (l%16) [+16 for subtile 1],
    // lane half picks d-range. All loads below use immediate offsets.
    const int drow = hi ? 16 : 0;
    const float* pk0 = bp + key0 + (long)mrow * strideElems + drow;
    const float* pk1 = pk0 + (long)16 * strideElems;

    // S = Q * K^T for two 16-key subtiles (bf16 WMMA, f32 accumulate).
    cfrag_t s0, s1;
    #pragma unroll
    for (int j = 0; j < 8; ++j) { s0[j] = 0.0f; s1[j] = 0.0f; }
    #pragma unroll
    for (int c = 0; c < 4; ++c) {
      bfrag_t kf0, kf1;
      #pragma unroll
      for (int e4 = 0; e4 < 4; ++e4) {
        f4_t a0 = *(const f4_t*)(pk0 + c * 32 + 4 * e4);
        f4_t a1 = *(const f4_t*)(pk1 + c * 32 + 4 * e4);
        #pragma unroll
        for (int h = 0; h < 4; ++h) {
          kf0[4 * e4 + h] = (__bf16)a0[h];
          kf1[4 * e4 + h] = (__bf16)a1[h];
        }
      }
      s0 = __builtin_amdgcn_wmma_f32_16x16x32_bf16(false, qa[c], false, kf0, (short)0, s0, false, false);
      s1 = __builtin_amdgcn_wmma_f32_16x16x32_bf16(false, qa[c], false, kf1, (short)0, s1, false, false);
    }

    // Online softmax over the 32 keys of this tile (rows live in C lanes 0-15).
    float alpha[GQ];
    #pragma unroll
    for (int g = 0; g < GQ; ++g) {
      float cm = fmaxf(s0[g], s1[g]);
      #pragma unroll
      for (int off = 8; off >= 1; off >>= 1)
        cm = fmaxf(cm, __shfl_xor(cm, off, 32));
      cm = wave_bcast0(cm);
      float mo = Mrun[g];
      float mn = fmaxf(mo, cm);
      alpha[g] = __expf(mo - mn);
      float p0v = __expf(s0[g] - mn);
      float p1v = __expf(s1[g] - mn);
      float ls  = p0v + p1v;
      #pragma unroll
      for (int off = 8; off >= 1; off >>= 1)
        ls += __shfl_xor(ls, off, 32);
      ls = wave_bcast0(ls);
      Mrun[g] = mn;
      Lrun[g] = Lrun[g] * alpha[g] + ls;
      if (!hi) {
        lds_p[wave][g][mrow]      = p0v;
        lds_p[wave][g][16 + mrow] = p1v;
      }
    }
    // Per-wave LDS round trip: DS is in-order per wave; wait for the stores.
    asm volatile("s_wait_dscnt 0" ::: "memory");

    // Rescale accumulator (rows >= GQ stay exactly 0).
    #pragma unroll
    for (int dt = 0; dt < 8; ++dt)
      #pragma unroll
      for (int v = 0; v < 8; ++v)
        oacc[dt][v] *= alpha[v & 3];

    // Repack P into A-fragment layout (rows >= GQ zero).
    bfrag_t pa;
    #pragma unroll
    for (int e = 0; e < 16; ++e) {
      int v = e >> 1, h = e & 1;
      int base = hi ? 8 : 0;
      int k = 2 * base + e;
      float val = (mrow < GQ) ? lds_p[wave][mrow][k] : 0.0f;
      pa[e] = (__bf16)val;
    }

    // V tile must be in LDS now.
    asm volatile("" ::: "memory");
    __builtin_amdgcn_s_wait_tensorcnt(0);
    asm volatile("" ::: "memory");

    // O += P * V over 8 d-tiles of 16. B-fragment gathers are ds_load_b32 with
    // immediate offsets; the TDM row pad (129 floats) makes them conflict-free.
    #pragma unroll
    for (int dt = 0; dt < 8; ++dt) {
      bfrag_t vf;
      #pragma unroll
      for (int e = 0; e < 16; ++e)
        vf[e] = (__bf16)vsp[vbase + e * VROW + dt * 16];
      oacc[dt] = __builtin_amdgcn_wmma_f32_16x16x32_bf16(false, pa, false, vf, (short)0, oacc[dt], false, false);
    }
  }

  // Publish per-wave partials.
  if (lane == 0) {
    #pragma unroll
    for (int g = 0; g < GQ; ++g) { lds_m[wave][g] = Mrun[g]; lds_l[wave][g] = Lrun[g]; }
  }
  if (!hi) {
    #pragma unroll
    for (int dt = 0; dt < 8; ++dt)
      #pragma unroll
      for (int g = 0; g < GQ; ++g)
        lds_o[wave][g][dt * 16 + mrow] = oacc[dt][g];
  }
  __syncthreads();

  // Merge the 4 waves + attention sink; thread tid handles dim d = tid.
  int d = tid;
  #pragma unroll
  for (int g = 0; g < GQ; ++g) {
    float sk = sink[kh * GQ + g];
    float mm = sk;
    for (int w = 0; w < NWAVES; ++w) mm = fmaxf(mm, lds_m[w][g]);
    float ll  = __expf(sk - mm);   // sink contributes only to the denominator
    float acc = 0.0f;
    for (int w = 0; w < NWAVES; ++w) {
      float f = __expf(lds_m[w][g] - mm);
      ll  += lds_l[w][g] * f;
      acc += lds_o[w][g][d] * f;
    }
    out[((b * HQn) + kh * GQ + g) * HEADD + d] = acc / ll;
  }
}

extern "C" void kernel_launch(void* const* d_in, const int* in_sizes, int n_in,
                              void* d_out, int out_size, void* d_ws, size_t ws_size,
                              hipStream_t stream) {
  const float* q    = (const float*)d_in[0];
  const float* raw  = (const float*)d_in[1];
  const float* sink = (const float*)d_in[2];
  const float* comp = (const float*)d_in[3];
  const int*   btab = (const int*)d_in[4];
  float* out = (float*)d_out;

  int B      = in_sizes[0] / (HQn * HEADD);                 // 64
  int nc     = in_sizes[3] / (B * 2 * HKV * HEADD);         // 96
  int nbps   = in_sizes[4] / B;                             // 32
  int winTok = nbps * 64 - WIN;                             // 1536

  dim3 grid(B * HKV), block(128);
  nsa_decode<<<grid, block, 0, stream>>>(q, raw, sink, comp, btab, out, nc, nbps, winTok);
}
